// DeepseekAttention_29635274342470
// MI455X (gfx1250) — compile-verified, matrix operand order checked by reading, not run on hardware
//
#include <hip/hip_runtime.h>
#include <hip/hip_bf16.h>

// ---------------------------------------------------------------------------
// MHA + RoPE for MI455X (gfx1250, wave32, WMMA 16x16x32 bf16)
//   B=1, S=2048, H=4096, 32 heads x 128 dim
// Pipeline:
//   1) gemm<MODE 0> X*Wq^T -> Qb  bf16 [head][s][d]
//      gemm<MODE 0> X*Wk^T -> Kb  bf16 [head][s][d]
//      gemm<MODE 1> X*Wv^T -> Vt  bf16 [head][d][s]   (transposed for PV matmul)
//   2) rope(Qb, scale=1/sqrt(128)), rope(Kb, scale=1)
//   3) flash attention (async global->LDS staging, ASYNCcnt) -> AO bf16 [s][4096]
//   4) gemm<MODE 2, A=bf16> AO*Wo^T -> d_out f32
// ---------------------------------------------------------------------------

typedef __attribute__((ext_vector_type(16))) __bf16 v16bf;
typedef __attribute__((ext_vector_type(8)))  float  v8f;

#define SEQ_LEN 2048
#define HID     4096
#define NHEADS  32
#define HDIM    128

__device__ __forceinline__ v8f wmma_bf16(v16bf a, v16bf b, v8f c) {
  // D(16x16,f32) = A(16x32,bf16) * B(32x16,bf16) + C
  return __builtin_amdgcn_wmma_f32_16x16x32_bf16(
      /*neg_a=*/false, a, /*neg_b=*/false, b,
      /*c_mod=*/(short)0, c, /*reuse_a=*/false, /*reuse_b=*/false);
}

// Load a 16x32 bf16 fragment from a row-major tile (A layout; B fragments are
// loaded from the [N][K] transposed operand with the identical pattern).
// ISA 16-bit A layout: lane = 16*(k-half) + m; per-lane elements
// 0..7 = k in [kh*8, kh*8+8), 8..15 = k in [16+kh*8, 16+kh*8+8).
__device__ __forceinline__ v16bf frag_ld(const __bf16* tile, int row0, int col0, int ld) {
  const int lane = threadIdx.x & 31;
  const int m  = lane & 15;
  const int kh = (lane >> 4) & 1;
  const __bf16* p = tile + (size_t)(row0 + m) * ld + col0 + kh * 8;
  v16bf f;
  reinterpret_cast<uint4*>(&f)[0] = *reinterpret_cast<const uint4*>(p);
  reinterpret_cast<uint4*>(&f)[1] = *reinterpret_cast<const uint4*>(p + 16);
  return f;
}

// Async global->LDS copy of 16 bytes (gfx1250; tracked with ASYNCcnt).
// ISA: the immediate offset applies to the LDS and the memory address alike,
// so one immediate offset advances BOTH addresses.
#define ASYNC_CP16(ldsaddr, gptr, imm_off)                                    \
  asm volatile("global_load_async_to_lds_b128 %0, %1, off offset:" #imm_off   \
               :: "v"(ldsaddr), "v"(gptr) : "memory")

// ---------------------------------------------------------------------------
// Tiled GEMM: C[M,N] = A[M,K] * B[N,K]^T   (M=SEQ, N=HID, K=HID)
// Block 128x128, BK=32, 256 threads = 8 waves in 2x4; wave tile 64x32.
// MODE 0: store bf16 [head][s][d]      (Q, K)
// MODE 1: store bf16 [head][d][s]      (V transposed)
// MODE 2: store f32  [s][o]            (final output projection)
// ---------------------------------------------------------------------------
template <int MODE, bool ABF16>
__global__ __launch_bounds__(256) void gemm_bf16_kernel(const void* __restrict__ Av,
                                                        const float* __restrict__ Bw,
                                                        void* __restrict__ Outv) {
  __shared__ __bf16 sA[2][128][40];   // +8 pad: spreads LDS banks, keeps 16B align
  __shared__ __bf16 sB[2][128][40];

  const int t    = threadIdx.x;
  const int m0   = blockIdx.y * 128;
  const int n0   = blockIdx.x * 128;
  const int tx   = t & 7;        // staging: 8 threads x 4 floats cover 32 cols
  const int ty   = t >> 3;       // 32 rows, x4 iterations -> 128 rows
  const int wave = t >> 5;
  const int lane = t & 31;
  const int wm   = wave >> 2;    // 0..1 -> 64-row wave tile
  const int wn   = wave & 3;     // 0..3 -> 32-col wave tile

  const float*  Af = (const float*)Av;
  const __bf16* Ab = (const __bf16*)Av;

  v8f acc[4][2] = {};

  auto stage = [&](int buf, int k0) {
#pragma unroll
    for (int r = 0; r < 4; ++r) {
      const int row = ty + (r << 5);
      __bf16* da = &sA[buf][row][tx * 4];
      if (ABF16) {
        *reinterpret_cast<uint2*>(da) =
            *reinterpret_cast<const uint2*>(Ab + (size_t)(m0 + row) * HID + k0 + tx * 4);
      } else {
        const float4 v =
            *reinterpret_cast<const float4*>(Af + (size_t)(m0 + row) * HID + k0 + tx * 4);
        __bf16 ta[4] = {(__bf16)v.x, (__bf16)v.y, (__bf16)v.z, (__bf16)v.w};
        *reinterpret_cast<uint2*>(da) = *reinterpret_cast<const uint2*>(ta);
      }
      const float4 w =
          *reinterpret_cast<const float4*>(Bw + (size_t)(n0 + row) * HID + k0 + tx * 4);
      __bf16 tb[4] = {(__bf16)w.x, (__bf16)w.y, (__bf16)w.z, (__bf16)w.w};
      *reinterpret_cast<uint2*>(&sB[buf][row][tx * 4]) = *reinterpret_cast<const uint2*>(tb);
    }
  };

  stage(0, 0);
  const int KT = HID / 32;
  int buf = 0;
  for (int kt = 0; kt < KT; ++kt) {
    __syncthreads();
    if (kt + 2 < KT) {  // gfx1250 global_prefetch_b8 for the weight stream
      __builtin_prefetch(Bw + (size_t)(n0 + ty) * HID + (kt + 2) * 32 + tx * 4, 0, 0);
    }
    if (kt + 1 < KT) stage(buf ^ 1, (kt + 1) * 32);

    const v16bf b0 = frag_ld(&sB[buf][0][0], wn * 32 + 0,  0, 40);
    const v16bf b1 = frag_ld(&sB[buf][0][0], wn * 32 + 16, 0, 40);
#pragma unroll
    for (int ms = 0; ms < 4; ++ms) {
      const v16bf a = frag_ld(&sA[buf][0][0], wm * 64 + ms * 16, 0, 40);
      acc[ms][0] = wmma_bf16(a, b0, acc[ms][0]);
      acc[ms][1] = wmma_bf16(a, b1, acc[ms][1]);
    }
    buf ^= 1;
  }

  // Epilogue. C layout: lane holds col n = lane%16; rows m = i + 8*(lane/16).
  const int h = lane >> 4;
#pragma unroll
  for (int ms = 0; ms < 4; ++ms) {
#pragma unroll
    for (int ns = 0; ns < 2; ++ns) {
      const int ncol = n0 + wn * 32 + ns * 16 + (lane & 15);
#pragma unroll
      for (int i = 0; i < 8; ++i) {
        const int mrow = m0 + wm * 64 + ms * 16 + i + 8 * h;
        const float val = acc[ms][ns][i];
        if (MODE == 2) {
          ((float*)Outv)[(size_t)mrow * HID + ncol] = val;
        } else {
          const int head = ncol >> 7;
          const int d    = ncol & 127;
          __bf16* O = (__bf16*)Outv;
          if (MODE == 0)
            O[((size_t)head * SEQ_LEN + mrow) * HDIM + d] = (__bf16)val;
          else
            O[((size_t)head * HDIM + d) * SEQ_LEN + mrow] = (__bf16)val;
        }
      }
    }
  }
}

// ---------------------------------------------------------------------------
// RoPE on [head][s][d] bf16; pairs (d, d+64); folds optional output scaling.
// ---------------------------------------------------------------------------
__global__ __launch_bounds__(256) void rope_kernel(__bf16* __restrict__ X,
                                                   const int* __restrict__ pos,
                                                   float outscale) {
  const int idx  = blockIdx.x * 256 + threadIdx.x;     // (head, s, d<64)
  const int d    = idx & 63;
  const int s    = (idx >> 6) & (SEQ_LEN - 1);
  const int head = idx >> 17;                          // /(64*2048)
  if (head >= NHEADS) return;
  __bf16* base = X + ((size_t)head * SEQ_LEN + s) * HDIM;
  const float x1 = (float)base[d];
  const float x2 = (float)base[d + 64];
  // inv_freq = theta^{-d/64}, theta = 10000
  const float invf = __expf(-(float)d * 0.015625f * 9.210340371976184f);
  const float ang  = (float)pos[s] * invf;
  float sn, c;
  __sincosf(ang, &sn, &c);
  base[d]      = (__bf16)((x1 * c - x2 * sn) * outscale);
  base[d + 64] = (__bf16)((x2 * c + x1 * sn) * outscale);
}

// ---------------------------------------------------------------------------
// Flash attention. Grid (SEQ/64, NHEADS), 128 threads (4 waves).
// Each wave owns a 16-query tile; 32-key tiles stream through double-buffered
// LDS via async global->LDS copies (ASYNCcnt), overlapped with WMMA compute.
// Q is pre-scaled by 1/sqrt(HDIM) in rope_kernel.
// ---------------------------------------------------------------------------
__global__ __launch_bounds__(128) void attn_kernel(const __bf16* __restrict__ Q,
                                                   const __bf16* __restrict__ K,
                                                   const __bf16* __restrict__ Vt,
                                                   __bf16* __restrict__ AO) {
  __shared__ __bf16 sK[2][32][136];   // [key][dim]
  __shared__ __bf16 sV[2][HDIM][40];  // [dim][key]  (from transposed V)
  __shared__ __bf16 sP[4][16][40];    // per-wave P scratch (C-layout -> A-layout)

  const int t    = threadIdx.x;
  const int wave = t >> 5;
  const int lane = t & 31;
  const int head = blockIdx.y;
  const int q0   = blockIdx.x * 64 + wave * 16;

  // Per-wave Q fragments: 16 rows x 128 dims = 4 A-fragments, kept in VGPRs.
  const __bf16* Qh = Q + ((size_t)head * SEQ_LEN + q0) * HDIM;
  v16bf qf[4];
#pragma unroll
  for (int kk = 0; kk < 4; ++kk) qf[kk] = frag_ld(Qh, 0, kk * 32, HDIM);

  v8f   o[8] = {};
  float mst[8], lst[8];
#pragma unroll
  for (int i = 0; i < 8; ++i) { mst[i] = -1e30f; lst[i] = 0.f; }

  const int ktr = t >> 2;          // K staging: 32 rows x (4 x 32 dims)
  const int ktc = (t & 3) * 32;
  const __bf16* Kh  = K  + ((size_t)head * SEQ_LEN + ktr) * HDIM + ktc;
  const __bf16* Vth = Vt + ((size_t)head * HDIM + t) * SEQ_LEN;

  // 8 async b128 copies per thread per tile: 4 for K (64B), 4 for Vt (64B).
  auto stage_async = [&](int buf, int k0) {
    const __bf16* ks = Kh + (size_t)k0 * HDIM;
    const unsigned lk = (unsigned)(uintptr_t)&sK[buf][ktr][ktc];
    ASYNC_CP16(lk, ks, 0);
    ASYNC_CP16(lk, ks, 16);
    ASYNC_CP16(lk, ks, 32);
    ASYNC_CP16(lk, ks, 48);
    const __bf16* vs = Vth + k0;
    const unsigned lv = (unsigned)(uintptr_t)&sV[buf][t][0];
    ASYNC_CP16(lv, vs, 0);
    ASYNC_CP16(lv, vs, 16);
    ASYNC_CP16(lv, vs, 32);
    ASYNC_CP16(lv, vs, 48);
  };

  const int NT = SEQ_LEN / 32;
  stage_async(0, 0);
  int buf = 0;
  for (int kt = 0; kt < NT; ++kt) {
    if (kt + 1 < NT) {
      stage_async(buf ^ 1, (kt + 1) * 32);
      // 8 newest async ops belong to the next tile; async loads complete in
      // order, so ASYNCcnt<=8 means this tile's 8 copies have landed in LDS.
      asm volatile("s_wait_asynccnt 0x8" ::: "memory");
    } else {
      asm volatile("s_wait_asynccnt 0x0" ::: "memory");
    }
    __syncthreads();

    // scores S(16x32) = Q(16x128) * K(32x128)^T : 2 column tiles x 4 k-steps
    v8f s0 = {}, s1 = {};
#pragma unroll
    for (int kk = 0; kk < 4; ++kk) {
      const v16bf b0 = frag_ld(&sK[buf][0][0], 0,  kk * 32, 136);
      const v16bf b1 = frag_ld(&sK[buf][0][0], 16, kk * 32, 136);
      s0 = wmma_bf16(qf[kk], b0, s0);
      s1 = wmma_bf16(qf[kk], b1, s1);
    }

    // online softmax; rows of a 16-lane half reduce with xor 1/2/4/8
    float p0[8], p1[8];
#pragma unroll
    for (int i = 0; i < 8; ++i) {
      float mx = fmaxf(s0[i], s1[i]);
      mx = fmaxf(mx, __shfl_xor(mx, 1, 32));
      mx = fmaxf(mx, __shfl_xor(mx, 2, 32));
      mx = fmaxf(mx, __shfl_xor(mx, 4, 32));
      mx = fmaxf(mx, __shfl_xor(mx, 8, 32));
      const float mnew = fmaxf(mst[i], mx);
      const float sc   = __expf(mst[i] - mnew);
      p0[i] = __expf(s0[i] - mnew);
      p1[i] = __expf(s1[i] - mnew);
      float rs = p0[i] + p1[i];
      rs += __shfl_xor(rs, 1, 32);
      rs += __shfl_xor(rs, 2, 32);
      rs += __shfl_xor(rs, 4, 32);
      rs += __shfl_xor(rs, 8, 32);
      lst[i] = lst[i] * sc + rs;
      mst[i] = mnew;
#pragma unroll
      for (int f = 0; f < 8; ++f) o[f][i] *= sc;
    }

    // C-layout -> A-layout via per-wave LDS scratch
    const int col = lane & 15;
    const int h   = lane >> 4;
#pragma unroll
    for (int i = 0; i < 8; ++i) {
      const int m = i + 8 * h;
      sP[wave][m][col]      = (__bf16)p0[i];
      sP[wave][m][col + 16] = (__bf16)p1[i];
    }
    const v16bf pa = frag_ld(&sP[wave][0][0], 0, 0, 40);
#pragma unroll
    for (int f = 0; f < 8; ++f) {
      const v16bf bv = frag_ld(&sV[buf][0][0], f * 16, 0, 40);
      o[f] = wmma_bf16(pa, bv, o[f]);
    }

    __syncthreads();   // all waves done with buf before it is re-staged
    buf ^= 1;
  }

  // normalize and write AO[s][head*128 + d] (bf16)
  const int col = lane & 15;
  const int h   = lane >> 4;
#pragma unroll
  for (int f = 0; f < 8; ++f) {
#pragma unroll
    for (int i = 0; i < 8; ++i) {
      const int m = i + 8 * h;
      const int s = q0 + m;
      const int d = f * 16 + col;
      AO[(size_t)s * HID + head * HDIM + d] = (__bf16)(o[f][i] / lst[i]);
    }
  }
}

// ---------------------------------------------------------------------------
extern "C" void kernel_launch(void* const* d_in, const int* in_sizes, int n_in,
                              void* d_out, int out_size, void* d_ws, size_t ws_size,
                              hipStream_t stream) {
  (void)in_sizes; (void)n_in; (void)out_size; (void)ws_size;

  const int*   positions = (const int*)d_in[0];
  const float* X         = (const float*)d_in[1];
  const float* Wq        = (const float*)d_in[2];
  const float* Wk        = (const float*)d_in[3];
  const float* Wv        = (const float*)d_in[4];
  const float* Wo        = (const float*)d_in[5];

  const size_t elems = (size_t)NHEADS * SEQ_LEN * HDIM;  // 8M per tensor
  __bf16* Qb = (__bf16*)d_ws;
  __bf16* Kb = Qb + elems;
  __bf16* Vt = Kb + elems;
  __bf16* AO = Vt + elems;

  const dim3 ggrid(HID / 128, SEQ_LEN / 128);   // 32 x 16
  gemm_bf16_kernel<0, false><<<ggrid, 256, 0, stream>>>((const void*)X, Wq, (void*)Qb);
  gemm_bf16_kernel<0, false><<<ggrid, 256, 0, stream>>>((const void*)X, Wk, (void*)Kb);
  gemm_bf16_kernel<1, false><<<ggrid, 256, 0, stream>>>((const void*)X, Wv, (void*)Vt);

  const int rope_blocks = (NHEADS * SEQ_LEN * 64) / 256;  // 16384
  rope_kernel<<<rope_blocks, 256, 0, stream>>>(Qb, positions, 0.08838834764831845f);
  rope_kernel<<<rope_blocks, 256, 0, stream>>>(Kb, positions, 1.0f);

  attn_kernel<<<dim3(SEQ_LEN / 64, NHEADS), 128, 0, stream>>>(Qb, Kb, Vt, AO);

  gemm_bf16_kernel<2, true><<<ggrid, 256, 0, stream>>>((const void*)AO, Wo, d_out);
}
